// DNN_23484881174931
// MI455X (gfx1250) — matrix operand order repaired and checked by reading, not compiled, not run on hardware
//
#include <hip/hip_runtime.h>
#include <hip/hip_bf16.h>

typedef __attribute__((ext_vector_type(16))) _Float16 v16h;
typedef __attribute__((ext_vector_type(8)))  _Float16 v8h;
typedef __attribute__((ext_vector_type(8)))  float    v8f;

#define MROWS   64          // rows per workgroup
#define THREADS 512         // 16 waves (wave32)
#define LDH     520         // padded LDS stride for h (512+8 halves)
#define KC      160         // layer-1 K chunk (5 WMMA k-steps)
#define LDAS    168         // padded LDS stride for staged x chunk

// f16 weight workspace layout (offsets in halves)
#define W1H_OFF 0                   // 512 x 800 (784 padded to 800 with zeros)
#define W2H_OFF 409600              // 512 x 512
#define W3H_OFF 671744              // 512 x 512
#define W4H_OFF 933888              // 10 x 512
#define WH_TOTAL 939008

static __device__ __forceinline__ v16h cat8(v8h a, v8h b) {
  return __builtin_shufflevector(a, b, 0,1,2,3,4,5,6,7,8,9,10,11,12,13,14,15);
}

static __device__ __forceinline__ v8f wmma_f16(v16h a, v16h b, v8f c) {
  // v_wmma_f32_16x16x32_f16: D = A(16x32 f16) x B(32x16 f16) + C(16x16 f32)
  return __builtin_amdgcn_wmma_f32_16x16x32_f16(false, a, false, b, (short)0, c,
                                                false, false);
}

// ---------------------------------------------------------------------------
// One-time (per launch) fp32 -> f16 weight conversion into workspace.
// W1 is padded along K from 784 to 800 with zeros so the WMMA K-loop is exact.
// ---------------------------------------------------------------------------
__global__ void convert_weights_kernel(const float* __restrict__ W1,
                                       const float* __restrict__ W2,
                                       const float* __restrict__ W3,
                                       const float* __restrict__ W4,
                                       _Float16* __restrict__ wh) {
  int i = blockIdx.x * blockDim.x + threadIdx.x;
  if (i >= WH_TOTAL) return;
  float v;
  if (i < W2H_OFF) {
    int r = i / 800;
    int k = i - r * 800;
    v = (k < 784) ? W1[r * 784 + k] : 0.0f;
  } else if (i < W3H_OFF) {
    v = W2[i - W2H_OFF];
  } else if (i < W4H_OFF) {
    v = W3[i - W3H_OFF];
  } else {
    v = W4[i - W4H_OFF];
  }
  wh[i] = (_Float16)v;
}

// ---------------------------------------------------------------------------
// 512->512 dense layer, fully in LDS.
// Each wave: all 4 M-slabs (64 rows) x 2 N-tiles -> 8 f32 accumulators.
// A fragments from LDS (ds_load_b128 x2), B fragments straight from L2
// (one contiguous 32B read per lane; W is [out][in] row-major = the
// contiguous-k B order used here (A matches): lane -> N column, 16 contiguous K values per lane).
// ---------------------------------------------------------------------------
static __device__ __forceinline__ void dense512(const _Float16* Hin,
                                                _Float16* Hout,
                                                const _Float16* __restrict__ Wh,
                                                const float* __restrict__ bias) {
  const int tid   = threadIdx.x;
  const int lane  = tid & 31;
  const int wid   = tid >> 5;           // 0..15
  const int lrow  = lane & 15;
  const int hi    = (lane >> 4) & 1;
  const int aKoff = hi ? 16 : 0;        // A: lanes 16-31 hold K 16..31
  const int bKoff = hi ? 16 : 0;        // B: lanes 16-31 hold K 16..31
  const int nt0   = wid * 2;            // first of 2 N-tiles

  v8f acc[8] = {};                      // [m*2+n]
  for (int ks = 0; ks < 16; ++ks) {
    const int k0 = ks * 32;
    v16h a[4];
#pragma unroll
    for (int m = 0; m < 4; ++m) {
      const _Float16* p = Hin + (m * 16 + lrow) * LDH + k0 + aKoff;
      v8h r0 = *(const v8h*)p;
      v8h r1 = *(const v8h*)(p + 8);
      a[m] = cat8(r0, r1);
    }
#pragma unroll
    for (int n = 0; n < 2; ++n) {
      const _Float16* wp =
          Wh + (size_t)((nt0 + n) * 16 + lrow) * 512 + k0 + bKoff;
      v16h b = *(const v16h*)wp;
#pragma unroll
      for (int m = 0; m < 4; ++m) acc[m * 2 + n] = wmma_f16(a[m], b, acc[m * 2 + n]);
    }
  }
  // Epilogue: bias + ReLU, write f16 tile back to LDS.
  // C/D layout: VGPR r -> row (m*16 + r + hi*8), col = ntile*16 + lrow.
#pragma unroll
  for (int n = 0; n < 2; ++n) {
    const int col = (nt0 + n) * 16 + lrow;
    const float bv = bias[col];
#pragma unroll
    for (int m = 0; m < 4; ++m) {
      v8f v = acc[m * 2 + n];
#pragma unroll
      for (int r = 0; r < 8; ++r) {
        float o = fmaxf(v[r] + bv, 0.0f);
        Hout[(m * 16 + r + hi * 8) * LDH + col] = (_Float16)o;
      }
    }
  }
}

// ---------------------------------------------------------------------------
// Fused MLP: x[64 rows] -> L1 -> L2 -> L3 -> L4 -> log_softmax, one WG pass.
// ---------------------------------------------------------------------------
__global__ __launch_bounds__(THREADS) void mlp_fused_kernel(
    const float* __restrict__ x,
    const float* __restrict__ b1, const float* __restrict__ b2,
    const float* __restrict__ b3, const float* __restrict__ b4,
    const _Float16* __restrict__ wh,
    float* __restrict__ out) {
  __shared__ __align__(32) _Float16 As[2][MROWS * LDAS];  // 43 KB (x chunks)
  __shared__ __align__(32) _Float16 H0[MROWS * LDH];      // 66.5 KB
  __shared__ __align__(32) _Float16 H1[MROWS * LDH];      // 66.5 KB
  __shared__ __align__(16) float    lg[MROWS * 10];       // 2.5 KB

  const int tid   = threadIdx.x;
  const int lane  = tid & 31;
  const int wid   = tid >> 5;
  const int lrow  = lane & 15;
  const int hi    = (lane >> 4) & 1;
  const int aKoff = hi ? 16 : 0;
  const int bKoff = hi ? 16 : 0;
  const int nt0   = wid * 2;
  const size_t row0 = (size_t)blockIdx.x * MROWS;

  const _Float16* W1h = wh + W1H_OFF;

  // ---------------- Layer 1: [64 x 800] x [800 -> 512] -------------------
  // Double-buffered K-chunk pipeline: convert fp32 x -> f16 into LDS
  // (global->VGPR->cvt->ds_store; conversion forces data through VGPRs,
  // so plain loads + non-temporal hint is the right path, not TDM).
  for (int i = tid; i < MROWS * KC; i += THREADS) {
    int r = i / KC, kl = i - r * KC;
    float v = (kl < 784) ? __builtin_nontemporal_load(&x[(row0 + r) * 784 + kl])
                         : 0.0f;
    As[0][r * LDAS + kl] = (_Float16)v;
  }
  __syncthreads();

  v8f acc[8] = {};
  for (int c = 0; c < 5; ++c) {
    if (c < 4) {  // prefetch next chunk into the other buffer
      const int cn = c + 1;
      _Float16* dst = As[cn & 1];
      for (int i = tid; i < MROWS * KC; i += THREADS) {
        int r = i / KC, kl = i - r * KC;
        int kg = cn * KC + kl;
        float v = (kg < 784)
                      ? __builtin_nontemporal_load(&x[(row0 + r) * 784 + kg])
                      : 0.0f;
        dst[r * LDAS + kl] = (_Float16)v;
      }
    }
    const _Float16* Ab = As[c & 1];
    for (int ks = 0; ks < 5; ++ks) {
      const int k0 = ks * 32;
      v16h a[4];
#pragma unroll
      for (int m = 0; m < 4; ++m) {
        const _Float16* p = Ab + (m * 16 + lrow) * LDAS + k0 + aKoff;
        v8h r0 = *(const v8h*)p;
        v8h r1 = *(const v8h*)(p + 8);
        a[m] = cat8(r0, r1);
      }
      const int kg0 = c * KC + k0;
#pragma unroll
      for (int n = 0; n < 2; ++n) {
        const _Float16* wp =
            W1h + (size_t)((nt0 + n) * 16 + lrow) * 800 + kg0 + bKoff;
        v16h b = *(const v16h*)wp;
#pragma unroll
        for (int m = 0; m < 4; ++m)
          acc[m * 2 + n] = wmma_f16(a[m], b, acc[m * 2 + n]);
      }
    }
    __syncthreads();
  }
  // L1 epilogue -> H0 (bias + ReLU)
#pragma unroll
  for (int n = 0; n < 2; ++n) {
    const int col = (nt0 + n) * 16 + lrow;
    const float bv = b1[col];
#pragma unroll
    for (int m = 0; m < 4; ++m) {
      v8f v = acc[m * 2 + n];
#pragma unroll
      for (int r = 0; r < 8; ++r) {
        float o = fmaxf(v[r] + bv, 0.0f);
        H0[(m * 16 + r + hi * 8) * LDH + col] = (_Float16)o;
      }
    }
  }
  __syncthreads();

  // ---------------- Layers 2 and 3 (512 -> 512, LDS ping-pong) -----------
  dense512(H0, H1, wh + W2H_OFF, b2);
  __syncthreads();
  dense512(H1, H0, wh + W3H_OFF, b3);
  __syncthreads();

  // ---------------- Layer 4 (512 -> 10) + log_softmax ---------------------
  // 0.5% of FLOPs: VALU with vectorized f16 dot products.
  const _Float16* W4h = wh + W4H_OFF;
  for (int o = tid; o < MROWS * 10; o += THREADS) {
    int row = o / 10, n = o - row * 10;
    const _Float16* hr = &H0[row * LDH];
    const _Float16* wr = W4h + n * 512;
    float s = b4[n];
    for (int k = 0; k < 512; k += 8) {
      v8h hv = *(const v8h*)(hr + k);
      v8h wv = *(const v8h*)(wr + k);
#pragma unroll
      for (int j = 0; j < 8; ++j) s += (float)hv[j] * (float)wv[j];
    }
    lg[o] = s;
  }
  __syncthreads();
  if (tid < MROWS) {
    const float* lr = &lg[tid * 10];
    float m = lr[0];
#pragma unroll
    for (int n = 1; n < 10; ++n) m = fmaxf(m, lr[n]);
    float s = 0.0f;
#pragma unroll
    for (int n = 0; n < 10; ++n) s += __expf(lr[n] - m);
    const float lse = m + __logf(s);
    float* op = out + (row0 + tid) * 10;
#pragma unroll
    for (int n = 0; n < 10; ++n)
      __builtin_nontemporal_store(lr[n] - lse, &op[n]);
  }
}

// ---------------------------------------------------------------------------
extern "C" void kernel_launch(void* const* d_in, const int* in_sizes, int n_in,
                              void* d_out, int out_size, void* d_ws,
                              size_t ws_size, hipStream_t stream) {
  const float* x  = (const float*)d_in[0];
  const float* W1 = (const float*)d_in[1];
  const float* b1 = (const float*)d_in[2];
  const float* W2 = (const float*)d_in[3];
  const float* b2 = (const float*)d_in[4];
  const float* W3 = (const float*)d_in[5];
  const float* b3 = (const float*)d_in[6];
  const float* W4 = (const float*)d_in[7];
  const float* b4 = (const float*)d_in[8];
  _Float16* wh = (_Float16*)d_ws;  // needs WH_TOTAL*2 ~ 1.9 MB of scratch
  float* out = (float*)d_out;

  convert_weights_kernel<<<(WH_TOTAL + 255) / 256, 256, 0, stream>>>(W1, W2, W3,
                                                                    W4, wh);
  const int nblocks = 65536 / MROWS;  // 1024 workgroups
  mlp_fused_kernel<<<nblocks, THREADS, 0, stream>>>(x, b1, b2, b3, b4, wh, out);
}
